// GAT_19189913878841
// MI455X (gfx1250) — matrix operand order repaired and checked by reading, not compiled, not run on hardware
//
#include <hip/hip_runtime.h>
#include <math.h>

#define F_IN     64
#define EDGE_DIM 16
#define HID      128
#define HEADS    4
#define CDIM     32
#define NCLS     40

typedef __attribute__((ext_vector_type(16))) _Float16 v16h;
typedef __attribute__((ext_vector_type(8)))  float    v8f;

// ---------------- helpers ----------------

__device__ __forceinline__ void atomicMaxF(float* addr, float val) {
  // Sign-aware monotonic-int trick; requires init to -inf. Correct under mixed-sign races.
  if (val >= 0.0f) atomicMax((int*)addr, __float_as_int(val));
  else             atomicMin((unsigned int*)addr, __float_as_uint(val));
}

__global__ void k_fill(float* __restrict__ p, float v, int n) {
  int i = blockIdx.x * blockDim.x + threadIdx.x;
  if (i < n) p[i] = v;
}

// ---------------- mean edge_attr per destination (self-loop fill) ----------------

__global__ void k_edge_mean_accum(const long long* __restrict__ dst,
                                  const float* __restrict__ ea,
                                  float* __restrict__ sum_ea,
                                  float* __restrict__ cnt, int E) {
  int idx = blockIdx.x * blockDim.x + threadIdx.x;
  if (idx >= E * EDGE_DIM) return;
  int k = idx >> 4, j = idx & 15;
  int d = (int)dst[k];
  atomicAdd(&sum_ea[d * EDGE_DIM + j], ea[idx]);
  if (j == 0) atomicAdd(&cnt[d], 1.0f);
}

__global__ void k_mean_fin(float* __restrict__ mean_ea, const float* __restrict__ cnt, int N) {
  int idx = blockIdx.x * blockDim.x + threadIdx.x;
  if (idx >= N * EDGE_DIM) return;
  int i = idx >> 4;
  mean_ea[idx] = mean_ea[idx] / fmaxf(cnt[i], 1.0f);
}

// ---------------- fold a_e into We : ve[h][d] = sum_c ae[h,c] * We[h*C+c, d] ----------------

__global__ void k_ve(const float* __restrict__ ae, const float* __restrict__ We,
                     float* __restrict__ ve) {
  int t = threadIdx.x;           // 64 threads: (h, d)
  if (t >= HEADS * EDGE_DIM) return;
  int h = t / EDGE_DIM, d = t % EDGE_DIM;
  float s = 0.0f;
  for (int c = 0; c < CDIM; ++c)
    s += ae[h * CDIM + c] * We[(size_t)(h * CDIM + c) * EDGE_DIM + d];
  ve[h * EDGE_DIM + d] = s;
}

// ---------------- WMMA GEMM: C[M,HID] = A[M,K] @ W[HID,K]^T ----------------
// 256-thread block = 8 waves. Block tile: 128 rows x 128 cols.
// Wave w owns rows [row0+16w, row0+16w+16) and ALL 8 column tiles:
// one A fragment feeds 8 WMMAs. The 8 B fragments (one per column tile) are
// built once per k-step (wave t builds tile t's fragment) and shared via LDS.

__global__ void k_gemm_wmma(const float* __restrict__ A, const float* __restrict__ W,
                            float* __restrict__ C, int M, int K) {
  __shared__ v16h lds_b[8][32];          // [col tile][lane] : 8 KB

  const int tid  = threadIdx.x;
  const int wave = tid >> 5;
  const int lane = tid & 31;
  const int row0 = blockIdx.x * 128 + wave * 16;

  const int mA    = lane & 15;
  const int kselA = (lane >> 4) * 8;     // A 16x32 f16 layout: lanes 16-31 hold K+8 pairs
  const int nB    = lane & 15;
  const int kselB = (lane >> 4) * 8;     // B 32x16 f16 layout: lanes 16-31 hold K+8 pairs (same as A)

  // clamp A row for edge blocks (all waves must stay in the loop for barriers;
  // out-of-range rows read valid memory, their stores are guarded below)
  int arow_idx = row0 + mA; if (arow_idx >= M) arow_idx = M - 1;
  const float* arow = A + (size_t)arow_idx * K;
  // wave w stages the B fragment of column tile w (W rows 16w .. 16w+15)
  const float* wrow = W + (size_t)(wave * 16 + nB) * K;

  v8f acc[8] = {};

  for (int kk = 0; kk < K; kk += 32) {
    // build this wave's B fragment for its column tile
    v16h bstage;
#pragma unroll
    for (int v = 0; v < 8; ++v) {
      int kb = kk + ((v & 4) ? 16 : 0) + kselB + (v & 3) * 2;
      bstage[2 * v]     = (_Float16)wrow[kb];
      bstage[2 * v + 1] = (_Float16)wrow[kb + 1];
    }
    __syncthreads();                     // WAR: previous iteration's LDS reads done
    lds_b[wave][lane] = bstage;

    // A fragment for this wave's 16-row slab
    v16h a;
#pragma unroll
    for (int v = 0; v < 8; ++v) {
      int ka = kk + ((v & 4) ? 16 : 0) + kselA + (v & 3) * 2;
      a[2 * v]     = (_Float16)arow[ka];
      a[2 * v + 1] = (_Float16)arow[ka + 1];
    }
    __syncthreads();                     // RAW: all B fragments staged

#pragma unroll
    for (int t = 0; t < 8; ++t) {
      v16h b = lds_b[t][lane];
      acc[t] = __builtin_amdgcn_wmma_f32_16x16x32_f16(
          /*neg_a=*/false, a, /*neg_b=*/false, b,
          /*c_mod=*/(short)0, acc[t], /*reuse_a=*/false, /*reuse_b=*/false);
    }
  }

  const int mo = row0 + (lane >> 4) * 8;   // D layout: VGPR g -> M = g (+8 for lanes 16-31)
#pragma unroll
  for (int t = 0; t < 8; ++t) {
    const int no = t * 16 + (lane & 15);
#pragma unroll
    for (int g = 0; g < 8; ++g)
      if (mo + g < M) C[(size_t)(mo + g) * HID + no] = acc[t][g];
  }
}

// ---------------- per-node attention dots: asd[i,h] = h_i,h . a_s[h] ----------------

__global__ void k_node_dots(const float* __restrict__ hlin,
                            const float* __restrict__ a_s, const float* __restrict__ a_d,
                            float* __restrict__ asd, float* __restrict__ addd, int N) {
  int i = blockIdx.x * blockDim.x + threadIdx.x;
  if (i >= N) return;
  const float* hr = hlin + (size_t)i * HID;
#pragma unroll
  for (int h = 0; h < HEADS; ++h) {
    float s = 0.0f, d = 0.0f;
    for (int c = 0; c < CDIM; ++c) {
      float v = hr[h * CDIM + c];
      s += v * a_s[h * CDIM + c];
      d += v * a_d[h * CDIM + c];
    }
    asd[i * HEADS + h]  = s;
    addd[i * HEADS + h] = d;
  }
}

// ---------------- pass 1: logits + segment max over destinations ----------------

__device__ __forceinline__ float dot16(const float* __restrict__ a,
                                       const float* __restrict__ b) {
  float s = 0.0f;
#pragma unroll
  for (int j = 0; j < EDGE_DIM; ++j) s += a[j] * b[j];
  return s;
}

__global__ void k_edge_alpha(const long long* __restrict__ src, const long long* __restrict__ dst,
                             const float* __restrict__ eattr, const float* __restrict__ ve,
                             const float* __restrict__ asd, const float* __restrict__ addd,
                             float* __restrict__ alphaE, float* __restrict__ amax, int E) {
  int k = blockIdx.x * blockDim.x + threadIdx.x;
  if (k >= E) return;
  int s = (int)src[k], d = (int)dst[k];
  const float* ea = eattr + (size_t)k * EDGE_DIM;
#pragma unroll
  for (int h = 0; h < HEADS; ++h) {
    float al = asd[s * HEADS + h] + addd[d * HEADS + h] + dot16(ea, ve + h * EDGE_DIM);
    al = (al > 0.0f) ? al : 0.2f * al;     // leaky_relu(0.2)
    alphaE[(size_t)k * HEADS + h] = al;
    atomicMaxF(&amax[d * HEADS + h], al);
  }
}

__global__ void k_self_alpha(const float* __restrict__ mean_ea, const float* __restrict__ ve,
                             const float* __restrict__ asd, const float* __restrict__ addd,
                             float* __restrict__ alphaS, float* __restrict__ amax, int N) {
  int i = blockIdx.x * blockDim.x + threadIdx.x;
  if (i >= N) return;
  const float* ea = mean_ea + (size_t)i * EDGE_DIM;
#pragma unroll
  for (int h = 0; h < HEADS; ++h) {
    float al = asd[i * HEADS + h] + addd[i * HEADS + h] + dot16(ea, ve + h * EDGE_DIM);
    al = (al > 0.0f) ? al : 0.2f * al;
    alphaS[i * HEADS + h] = al;
    atomicMaxF(&amax[i * HEADS + h], al);
  }
}

// ---------------- pass 2: exp + segment sum (denominator) ----------------

__global__ void k_edge_expden(const long long* __restrict__ dst,
                              float* __restrict__ alphaE, const float* __restrict__ amax,
                              float* __restrict__ den, int E) {
  int k = blockIdx.x * blockDim.x + threadIdx.x;
  if (k >= E) return;
  int d = (int)dst[k];
#pragma unroll
  for (int h = 0; h < HEADS; ++h) {
    float ex = __expf(alphaE[(size_t)k * HEADS + h] - amax[d * HEADS + h]);
    alphaE[(size_t)k * HEADS + h] = ex;
    atomicAdd(&den[d * HEADS + h], ex);
  }
}

__global__ void k_self_expden(float* __restrict__ alphaS, const float* __restrict__ amax,
                              float* __restrict__ den, int N) {
  int i = blockIdx.x * blockDim.x + threadIdx.x;
  if (i >= N) return;
#pragma unroll
  for (int h = 0; h < HEADS; ++h) {
    float ex = __expf(alphaS[i * HEADS + h] - amax[i * HEADS + h]);
    alphaS[i * HEADS + h] = ex;
    den[i * HEADS + h] += ex;     // unique i per thread; edge pass already finished
  }
}

// ---------------- pass 3: weighted scatter-add (one wave per edge) ----------------

__global__ void k_edge_agg(const long long* __restrict__ src, const long long* __restrict__ dst,
                           const float* __restrict__ hlin, const float* __restrict__ alphaE,
                           const float* __restrict__ den, float* __restrict__ outb, int E) {
  int e = blockIdx.x * 8 + (threadIdx.x >> 5);   // 8 waves per 256-thread block
  if (e >= E) return;
  int lane = threadIdx.x & 31;
  int s = (int)src[e], d = (int)dst[e];
  const float* hs = hlin + (size_t)s * HID;
  float*       od = outb + (size_t)d * HID;
#pragma unroll
  for (int h = 0; h < HEADS; ++h) {
    float coef = alphaE[(size_t)e * HEADS + h] / (den[d * HEADS + h] + 1e-16f);
    atomicAdd(&od[h * CDIM + lane], hs[h * CDIM + lane] * coef);
  }
}

__global__ void k_self_agg(const float* __restrict__ hlin, const float* __restrict__ alphaS,
                           const float* __restrict__ den, float* __restrict__ outb, int N) {
  int idx = blockIdx.x * blockDim.x + threadIdx.x;
  if (idx >= N * HID) return;
  int i = idx / HID, c = idx % HID, h = c >> 5;
  float coef = alphaS[i * HEADS + h] / (den[i * HEADS + h] + 1e-16f);
  outb[idx] += hlin[idx] * coef;   // runs after edge agg kernel (stream-ordered)
}

__global__ void k_bias_relu(float* __restrict__ buf, const float* __restrict__ bias, int N) {
  int idx = blockIdx.x * blockDim.x + threadIdx.x;
  if (idx >= N * HID) return;
  buf[idx] = fmaxf(buf[idx] + bias[idx % HID], 0.0f);
}

// ---------------- classifier + log_softmax ----------------

__global__ void k_classifier(const float* __restrict__ hfeat, const float* __restrict__ Wlin,
                             const float* __restrict__ blin, float* __restrict__ out, int N) {
  int i = blockIdx.x * blockDim.x + threadIdx.x;
  if (i >= N) return;
  const float* hr = hfeat + (size_t)i * HID;
  float lg[NCLS];
  float mx = -INFINITY;
  for (int c = 0; c < NCLS; ++c) {
    float s = blin[c];
    const float* wr = Wlin + (size_t)c * HID;
    for (int j = 0; j < HID; ++j) s += hr[j] * wr[j];
    lg[c] = s;
    mx = fmaxf(mx, s);
  }
  float se = 0.0f;
  for (int c = 0; c < NCLS; ++c) se += __expf(lg[c] - mx);
  float lse = mx + __logf(se);
  for (int c = 0; c < NCLS; ++c) out[(size_t)i * NCLS + c] = lg[c] - lse;
}

// ---------------- host orchestration ----------------

extern "C" void kernel_launch(void* const* d_in, const int* in_sizes, int n_in,
                              void* d_out, int out_size, void* d_ws, size_t ws_size,
                              hipStream_t stream) {
  const float*     x     = (const float*)d_in[0];
  const long long* ei    = (const long long*)d_in[1];
  const float*     eattr = (const float*)d_in[2];
  const float*     W1    = (const float*)d_in[3];
  const float*     as1   = (const float*)d_in[4];
  const float*     ad1   = (const float*)d_in[5];
  const float*     ae1   = (const float*)d_in[6];
  const float*     We1   = (const float*)d_in[7];
  const float*     b1    = (const float*)d_in[8];
  const float*     W2    = (const float*)d_in[9];
  const float*     as2   = (const float*)d_in[10];
  const float*     ad2   = (const float*)d_in[11];
  const float*     ae2   = (const float*)d_in[12];
  const float*     We2   = (const float*)d_in[13];
  const float*     b2    = (const float*)d_in[14];
  const float*     Wlin  = (const float*)d_in[15];
  const float*     blin  = (const float*)d_in[16];

  const int N = in_sizes[0] / F_IN;       // 50000
  const int E = in_sizes[2] / EDGE_DIM;   // 1600000
  const long long* src = ei;
  const long long* dstv = ei + E;

  // workspace carve-out
  char* base = (char*)d_ws;
  size_t off = 0;
  auto alloc = [&](size_t nbytes) -> void* {
    off = (off + 255) & ~(size_t)255;
    void* p = base + off;
    off += nbytes;
    return p;
  };
  float* cnt    = (float*)alloc((size_t)N * 4);
  float* meanEA = (float*)alloc((size_t)N * EDGE_DIM * 4);
  float* ve     = (float*)alloc((size_t)HEADS * EDGE_DIM * 4);
  float* bufX   = (float*)alloc((size_t)N * HID * 4);
  float* bufY   = (float*)alloc((size_t)N * HID * 4);
  float* asd    = (float*)alloc((size_t)N * HEADS * 4);
  float* addd   = (float*)alloc((size_t)N * HEADS * 4);
  float* amax   = (float*)alloc((size_t)N * HEADS * 4);
  float* den    = (float*)alloc((size_t)N * HEADS * 4);
  float* alphaE = (float*)alloc((size_t)E * HEADS * 4);
  float* alphaS = (float*)alloc((size_t)N * HEADS * 4);
  (void)ws_size; (void)n_in; (void)out_size;

  auto blocks = [](long long n, int b) { return dim3((unsigned)((n + b - 1) / b)); };

  // self-loop edge_attr = per-destination mean of incoming edge_attr
  k_fill<<<blocks((long long)N * EDGE_DIM, 256), 256, 0, stream>>>(meanEA, 0.0f, N * EDGE_DIM);
  k_fill<<<blocks(N, 256), 256, 0, stream>>>(cnt, 0.0f, N);
  k_edge_mean_accum<<<blocks((long long)E * EDGE_DIM, 256), 256, 0, stream>>>(dstv, eattr, meanEA, cnt, E);
  k_mean_fin<<<blocks((long long)N * EDGE_DIM, 256), 256, 0, stream>>>(meanEA, cnt, N);

  auto run_layer = [&](const float* feat_in, int K,
                       const float* Wm, const float* a_s, const float* a_d, const float* a_e,
                       const float* We, const float* bias, float* lin, float* aggout) {
    k_ve<<<1, 64, 0, stream>>>(a_e, We, ve);
    k_gemm_wmma<<<dim3((unsigned)((N + 127) / 128)), 256, 0, stream>>>(feat_in, Wm, lin, N, K);
    k_node_dots<<<blocks(N, 256), 256, 0, stream>>>(lin, a_s, a_d, asd, addd, N);
    k_fill<<<blocks((long long)N * HEADS, 256), 256, 0, stream>>>(amax, -INFINITY, N * HEADS);
    k_fill<<<blocks((long long)N * HEADS, 256), 256, 0, stream>>>(den, 0.0f, N * HEADS);
    k_fill<<<blocks((long long)N * HID, 256), 256, 0, stream>>>(aggout, 0.0f, N * HID);
    k_edge_alpha<<<blocks(E, 256), 256, 0, stream>>>(src, dstv, eattr, ve, asd, addd, alphaE, amax, E);
    k_self_alpha<<<blocks(N, 256), 256, 0, stream>>>(meanEA, ve, asd, addd, alphaS, amax, N);
    k_edge_expden<<<blocks(E, 256), 256, 0, stream>>>(dstv, alphaE, amax, den, E);
    k_self_expden<<<blocks(N, 256), 256, 0, stream>>>(alphaS, amax, den, N);
    k_edge_agg<<<dim3((unsigned)((E + 7) / 8)), 256, 0, stream>>>(src, dstv, lin, alphaE, den, aggout, E);
    k_self_agg<<<blocks((long long)N * HID, 256), 256, 0, stream>>>(lin, alphaS, den, aggout, N);
    k_bias_relu<<<blocks((long long)N * HID, 256), 256, 0, stream>>>(aggout, bias, N);
  };

  // layer 1: x -> bufX (lin), aggregate -> bufY (becomes hfeat1 after bias+relu)
  run_layer(x, F_IN, W1, as1, ad1, ae1, We1, b1, bufX, bufY);
  // layer 2: hfeat1 (bufY) -> bufX (lin), aggregate -> bufY (hfeat2)
  run_layer(bufY, HID, W2, as2, ad2, ae2, We2, b2, bufX, bufY);

  // classifier + log_softmax
  k_classifier<<<blocks(N, 256), 256, 0, stream>>>(bufY, Wlin, blin, (float*)d_out, N);
}
